// ImprovedADHDHybridGAT_68968584839587
// MI455X (gfx1250) — matrix operand order repaired and checked by reading, not compiled, not run on hardware
//
#include <hip/hip_runtime.h>
#include <hip/hip_bf16.h>
#include <cstdint>

// ---------------------------------------------------------------------------
// Model constants (match reference)
// ---------------------------------------------------------------------------
#define NN   8192      // nodes
#define EE   131072    // edges
#define GG   32        // graphs
#define FIN  256
#define HH   8
#define CC   128
#define HCC  1024
#define PP   64
#define ENED (EE + NN) // edges + self loops

typedef __attribute__((ext_vector_type(16))) _Float16 v16h;
typedef __attribute__((ext_vector_type(8)))  _Float16 h8;
typedef __attribute__((ext_vector_type(8)))  float    v8f;

// monotonic float<->uint encoding for atomic max on possibly-negative floats
__device__ __forceinline__ unsigned fenc(float f) {
    unsigned u = __float_as_uint(f);
    return (u & 0x80000000u) ? ~u : (u | 0x80000000u);
}
__device__ __forceinline__ float fdec(unsigned u) {
    return (u & 0x80000000u) ? __uint_as_float(u & 0x7FFFFFFFu) : __uint_as_float(~u);
}
#define FENC_NEG_INF 0x007FFFFFu   // fenc(-inf)

// ---------------------------------------------------------------------------
// Utility kernels
// ---------------------------------------------------------------------------
__global__ void k_fill_u32(unsigned* p, unsigned v, size_t n) {
    size_t i = (size_t)blockIdx.x * blockDim.x + threadIdx.x;
    if (i < n) p[i] = v;
}

__global__ void k_cast_f16(const float* __restrict__ x, _Float16* __restrict__ y, size_t n) {
    size_t i = (size_t)blockIdx.x * blockDim.x + threadIdx.x;
    if (i < n) y[i] = (_Float16)x[i];
}

// W [K,N] f32 row-major  ->  Wt [N,K] f16 row-major
__global__ void k_cast_transpose(const float* __restrict__ W, _Float16* __restrict__ Wt,
                                 int K, int N) {
    int i = blockIdx.x * blockDim.x + threadIdx.x;
    if (i >= K * N) return;
    int k = i / N, n = i - k * N;
    Wt[(size_t)n * K + k] = (_Float16)W[i];
}

// build src/dst int arrays with PyG-style self loops appended
__global__ void k_build_edges(const int* __restrict__ ei, int* __restrict__ src,
                              int* __restrict__ dst) {
    int i = blockIdx.x * blockDim.x + threadIdx.x;
    if (i < EE)            { src[i] = ei[i]; dst[i] = ei[EE + i]; }
    else if (i < ENED)     { src[i] = i - EE; dst[i] = i - EE; }
}

// ---------------------------------------------------------------------------
// WMMA GEMM:  C[M,N] = A[M,K] * B[K,N],  A f16 row-major, Bt = B^T f16 [N,K].
// One wave computes a 16x64 tile: one A fragment feeds 4 WMMAs per k-step
// (4x A reuse in registers); the block's 4 waves share the same 4 B fragments
// through L0. Fragments use contiguous k order (k = 16*half + s) in both A and B.
// ---------------------------------------------------------------------------
#define GEMM_WAVES 4
#define GEMM_NT    4            // N tiles (of 16) per wave -> 64 columns
__global__ void __launch_bounds__(GEMM_WAVES * 32)
k_gemm_f16_wmma(const _Float16* __restrict__ A, const _Float16* __restrict__ Bt,
                float* __restrict__ C, int M, int N, int K) {
    const int lane = threadIdx.x & 31;
    const int wave = threadIdx.x >> 5;
    const int tn0 = blockIdx.x * GEMM_NT;             // first of 4 N tiles
    const int tm  = blockIdx.y * GEMM_WAVES + wave;
    const int r   = lane & 15;
    const int hs  = lane >> 4;                        // lane half select
    // A fragment: lane holds row (tm*16+r); 16 contiguous K values at k+hs*16
    const _Float16* arow = A  + (size_t)(tm * 16 + r) * K + hs * 16;
    // B fragments: lane holds column; 16 contiguous K values at k+hs*16
    const _Float16* brow = Bt + (size_t)(tn0 * 16 + r) * K + hs * 16;
    const size_t bstr = (size_t)16 * K;               // halfs between N tiles
    v8f acc0 = {0.f,0.f,0.f,0.f,0.f,0.f,0.f,0.f};
    v8f acc1 = acc0, acc2 = acc0, acc3 = acc0;
    for (int k = 0; k < K; k += 32) {
        v16h a = *(const v16h*)(arow + k);
        v16h b0 = *(const v16h*)(brow + k);
        v16h b1 = *(const v16h*)(brow + bstr + k);
        v16h b2 = *(const v16h*)(brow + 2 * bstr + k);
        v16h b3 = *(const v16h*)(brow + 3 * bstr + k);
        acc0 = __builtin_amdgcn_wmma_f32_16x16x32_f16(false, a, false, b0, (short)0, acc0, false, false);
        acc1 = __builtin_amdgcn_wmma_f32_16x16x32_f16(false, a, false, b1, (short)0, acc1, false, false);
        acc2 = __builtin_amdgcn_wmma_f32_16x16x32_f16(false, a, false, b2, (short)0, acc2, false, false);
        acc3 = __builtin_amdgcn_wmma_f32_16x16x32_f16(false, a, false, b3, (short)0, acc3, false, false);
    }
    // C/D layout: VGPR i -> row hs*8+i, col = lane&15 within each 16-col tile
    float* cbase = C + (size_t)(tm * 16 + hs * 8) * N + tn0 * 16 + r;
#pragma unroll
    for (int i = 0; i < 8; ++i) {
        float* cr = cbase + (size_t)i * N;
        cr[0]  = acc0[i];
        cr[16] = acc1[i];
        cr[32] = acc2[i];
        cr[48] = acc3[i];
    }
}

// ---------------------------------------------------------------------------
// Attention coefficients: als[n,h] = sum_c xl[n,h,c]*a_src[h,c]  (warp per head)
// ---------------------------------------------------------------------------
__global__ void k_attn_coef(const float* __restrict__ xl, const float* __restrict__ a_src,
                            const float* __restrict__ a_dst, float* __restrict__ als,
                            float* __restrict__ ald, int H) {
    int n = blockIdx.x;
    int h = threadIdx.x >> 5, lane = threadIdx.x & 31;
    const float* v  = xl + (size_t)n * H * CC + (size_t)h * CC;
    const float* as = a_src + (size_t)h * CC;
    const float* ad = a_dst + (size_t)h * CC;
    float ss = 0.f, sd = 0.f;
    for (int c = lane; c < CC; c += 32) { float x = v[c]; ss += x * as[c]; sd += x * ad[c]; }
#pragma unroll
    for (int o = 16; o > 0; o >>= 1) { ss += __shfl_down(ss, o); sd += __shfl_down(sd, o); }
    if (lane == 0) { als[n * H + h] = ss; ald[n * H + h] = sd; }
}

// ---------------------------------------------------------------------------
// Edge segment-softmax passes
// ---------------------------------------------------------------------------
__global__ void k_edge_logits(const int* __restrict__ src, const int* __restrict__ dst,
                              const float* __restrict__ als, const float* __restrict__ ald,
                              float* __restrict__ elog, unsigned* __restrict__ maxb, int H) {
    int i = blockIdx.x * blockDim.x + threadIdx.x;
    if (i >= ENED * H) return;
    int e = i / H, h = i - e * H;
    int s = src[e], d = dst[e];
    float l = als[s * H + h] + ald[d * H + h];
    l = (l > 0.f) ? l : 0.2f * l;                       // leaky_relu(0.2)
    elog[i] = l;
    atomicMax(&maxb[d * H + h], fenc(l));
}

__global__ void k_edge_exp(const int* __restrict__ dst, float* __restrict__ elog,
                           const unsigned* __restrict__ maxb, float* __restrict__ sumb, int H) {
    int i = blockIdx.x * blockDim.x + threadIdx.x;
    if (i >= ENED * H) return;
    int e = i / H, h = i - e * H;
    int d = dst[e];
    float ex = __expf(elog[i] - fdec(maxb[d * H + h]));
    elog[i] = ex;
    atomicAdd(&sumb[d * H + h], ex);
}

__global__ void k_edge_alpha(const int* __restrict__ dst, float* __restrict__ elog,
                             const float* __restrict__ sumb, int H) {
    int i = blockIdx.x * blockDim.x + threadIdx.x;
    if (i >= ENED * H) return;
    int e = i / H, h = i - e * H;
    elog[i] = elog[i] / sumb[dst[e] * H + h];
}

// out[d,h,:] += alpha[e,h] * xl[s,h,:]  (block per edge)
__global__ void k_aggregate(const int* __restrict__ src, const int* __restrict__ dst,
                            const float* __restrict__ alpha, const float* __restrict__ xl,
                            float* __restrict__ agg, int H) {
    int e = blockIdx.x;
    int s = src[e], d = dst[e];
    int HC = H * CC;
    const float* xs = xl + (size_t)s * HC;
    float* od = agg + (size_t)d * HC;
    for (int idx = threadIdx.x; idx < HC; idx += blockDim.x) {
        float a = alpha[e * H + (idx >> 7)];            // idx / CC, CC=128
        atomicAdd(&od[idx], a * xs[idx]);
    }
}

// y = elu((agg + b) * g + be); emit f16 (next GEMM input) and optional f32
__global__ void k_post(const float* __restrict__ agg, const float* __restrict__ b,
                       const float* __restrict__ g, const float* __restrict__ be,
                       float* __restrict__ out32, _Float16* __restrict__ out16,
                       int HC, size_t total) {
    size_t i = (size_t)blockIdx.x * blockDim.x + threadIdx.x;
    if (i >= total) return;
    int c = (int)(i % HC);
    float v = (agg[i] + b[c]) * g[c] + be[c];
    v = (v > 0.f) ? v : (__expf(v) - 1.f);              // elu
    out16[i] = (_Float16)v;
    if (out32) out32[i] = v;
}

// ---------------------------------------------------------------------------
// Graph pooling (mean + max)
// ---------------------------------------------------------------------------
__global__ void k_pool_acc(const float* __restrict__ x3, const int* __restrict__ batch,
                           float* __restrict__ psum, unsigned* __restrict__ pmax,
                           float* __restrict__ pcnt) {
    int n = blockIdx.x;
    int g = batch[n];
    const float* xr = x3 + (size_t)n * CC;
    for (int c = threadIdx.x; c < CC; c += blockDim.x) {
        atomicAdd(&psum[g * CC + c], xr[c]);
        atomicMax(&pmax[g * CC + c], fenc(xr[c]));
    }
    if (threadIdx.x == 0) atomicAdd(&pcnt[g], 1.f);
}

__global__ void k_pool_fin(const float* __restrict__ psum, const unsigned* __restrict__ pmax,
                           const float* __restrict__ pcnt, float* __restrict__ xg) {
    int i = blockIdx.x * blockDim.x + threadIdx.x;
    if (i >= GG * CC) return;
    int g = i >> 7;
    float c = pcnt[g];
    float mean = psum[i] / fmaxf(c, 1.f);
    float mx = (c > 0.f) ? fdec(pmax[i]) : 0.f;
    xg[i] = 0.5f * (mean + mx);
}

// ---------------------------------------------------------------------------
// Pheno MLP + fusion attention + classifier (one block per graph)
// ---------------------------------------------------------------------------
__global__ void __launch_bounds__(160)
k_head(const float* __restrict__ xg, const float* __restrict__ pheno,
       const float* Wp1, const float* bp1, const float* gp1, const float* bep1,
       const float* Wp2, const float* bp2, const float* gp2, const float* bep2,
       const float* Wp3, const float* bp3,
       const float* Wa1, const float* ba1, const float* Wa2, const float* ba2,
       const float* Wc1, const float* bc1, const float* gc1, const float* bec1,
       const float* Wc2, const float* bc2, const float* Wc3, const float* bc3,
       float* __restrict__ out) {
    int g = blockIdx.x, t = threadIdx.x;
    __shared__ float h1[64], h2[32], comb[144], tnh[144], c1[64], c2[32], attw;
    const float* ph = pheno + (size_t)g * PP;
    if (t < 64) {
        float s = bp1[t];
        for (int k = 0; k < 64; ++k) s += ph[k] * Wp1[k * 64 + t];
        s = s * gp1[t] + bep1[t];
        h1[t] = fmaxf(s, 0.f);
    }
    __syncthreads();
    if (t < 32) {
        float s = bp2[t];
        for (int k = 0; k < 64; ++k) s += h1[k] * Wp2[k * 32 + t];
        s = s * gp2[t] + bep2[t];
        h2[t] = fmaxf(s, 0.f);
    }
    __syncthreads();
    if (t < 16) {
        float s = bp3[t];
        for (int k = 0; k < 32; ++k) s += h2[k] * Wp3[k * 16 + t];
        comb[128 + t] = fmaxf(s, 0.f);
    }
    if (t < 128) comb[t] = xg[(size_t)g * CC + t];
    __syncthreads();
    if (t < 144) {
        float s = ba1[t];
        for (int k = 0; k < 144; ++k) s += comb[k] * Wa1[k * 144 + t];
        tnh[t] = tanhf(s);
    }
    __syncthreads();
    if (t == 0) {
        float s = ba2[0];
        for (int k = 0; k < 144; ++k) s += tnh[k] * Wa2[k];
        attw = 1.f / (1.f + __expf(-s));
    }
    __syncthreads();
    if (t < 64) {
        float s = bc1[t];
        for (int k = 0; k < 144; ++k) s += comb[k] * attw * Wc1[k * 64 + t];
        s = s * gc1[t] + bec1[t];
        c1[t] = fmaxf(s, 0.f);
    }
    __syncthreads();
    if (t < 32) {
        float s = bc2[t];
        for (int k = 0; k < 64; ++k) s += c1[k] * Wc2[k * 32 + t];
        c2[t] = fmaxf(s, 0.f);
    }
    __syncthreads();
    if (t < 2) {
        float s = bc3[t];
        for (int k = 0; k < 32; ++k) s += c2[k] * Wc3[k * 2 + t];
        out[g * 2 + t] = s;
    }
}

// ---------------------------------------------------------------------------
// Host orchestration
// ---------------------------------------------------------------------------
extern "C" void kernel_launch(void* const* d_in, const int* in_sizes, int n_in,
                              void* d_out, int out_size, void* d_ws, size_t ws_size,
                              hipStream_t stream) {
    (void)in_sizes; (void)n_in; (void)out_size; (void)ws_size;
    const float* x     = (const float*)d_in[0];
    const int*   eidx  = (const int*)d_in[1];
    const int*   batch = (const int*)d_in[2];
    const float* pheno = (const float*)d_in[3];
    const float* Wl[3]  = {(const float*)d_in[4],  (const float*)d_in[10], (const float*)d_in[16]};
    const float* asl[3] = {(const float*)d_in[5],  (const float*)d_in[11], (const float*)d_in[17]};
    const float* adl[3] = {(const float*)d_in[6],  (const float*)d_in[12], (const float*)d_in[18]};
    const float* bl[3]  = {(const float*)d_in[7],  (const float*)d_in[13], (const float*)d_in[19]};
    const float* gl[3]  = {(const float*)d_in[8],  (const float*)d_in[14], (const float*)d_in[20]};
    const float* bel[3] = {(const float*)d_in[9],  (const float*)d_in[15], (const float*)d_in[21]};
    float* out = (float*)d_out;

    // workspace carve (256B aligned blocks)
    char* wp = (char*)d_ws;
    auto carve = [&](size_t bytes) -> char* {
        char* p = wp; wp += (bytes + 255) & ~(size_t)255; return p;
    };
    _Float16* XH   = (_Float16*)carve((size_t)NN * HCC * 2);   // f16 activations (GEMM A)
    _Float16* WTH  = (_Float16*)carve((size_t)HCC * HCC * 2);  // f16 transposed weights (GEMM B)
    float*    XL   = (float*)carve((size_t)NN * HCC * 4);      // GEMM output xl
    float*    AGG  = (float*)carve((size_t)NN * HCC * 4);      // aggregated messages
    float*    ALS  = (float*)carve((size_t)NN * HH * 4);
    float*    ALD  = (float*)carve((size_t)NN * HH * 4);
    unsigned* MAXB = (unsigned*)carve((size_t)NN * HH * 4);
    float*    SUMB = (float*)carve((size_t)NN * HH * 4);
    float*    ELOG = (float*)carve((size_t)ENED * HH * 4);
    int*      SRC  = (int*)carve((size_t)ENED * 4);
    int*      DST  = (int*)carve((size_t)ENED * 4);
    float*    X3   = (float*)carve((size_t)NN * CC * 4);
    float*    PSUM = (float*)carve((size_t)GG * CC * 4);
    unsigned* PMAX = (unsigned*)carve((size_t)GG * CC * 4);
    float*    PCNT = (float*)carve((size_t)GG * 4);
    float*    XG   = (float*)carve((size_t)GG * CC * 4);

    const int Kin[3]  = {FIN, HCC, HCC};
    const int Nout[3] = {HCC, HCC, CC};
    const int Hh[3]   = {HH, HH, 1};

    // edges with self loops
    k_build_edges<<<(ENED + 255) / 256, 256, 0, stream>>>(eidx, SRC, DST);
    // x -> f16
    {
        size_t n = (size_t)NN * FIN;
        k_cast_f16<<<(unsigned)((n + 255) / 256), 256, 0, stream>>>(x, XH, n);
    }

    for (int l = 0; l < 3; ++l) {
        int K = Kin[l], N = Nout[l], H = Hh[l];
        size_t MN = (size_t)NN * N;
        // weights -> f16, transposed [N,K]
        k_cast_transpose<<<(K * N + 255) / 256, 256, 0, stream>>>(Wl[l], WTH, K, N);
        // xl = XH @ W  (WMMA, 16x64 tile per wave)
        dim3 gg(N / (16 * GEMM_NT), NN / (16 * GEMM_WAVES));
        k_gemm_f16_wmma<<<gg, GEMM_WAVES * 32, 0, stream>>>(XH, WTH, XL, NN, N, K);
        // attention coefficients
        k_attn_coef<<<NN, H * 32, 0, stream>>>(XL, asl[l], adl[l], ALS, ALD, H);
        // segment softmax over in-edges per dst
        size_t nh = (size_t)NN * H;
        k_fill_u32<<<(unsigned)((nh + 255) / 256), 256, 0, stream>>>(MAXB, FENC_NEG_INF, nh);
        k_fill_u32<<<(unsigned)((nh + 255) / 256), 256, 0, stream>>>((unsigned*)SUMB, 0u, nh);
        int enh_blocks = (ENED * H + 255) / 256;
        k_edge_logits<<<enh_blocks, 256, 0, stream>>>(SRC, DST, ALS, ALD, ELOG, MAXB, H);
        k_edge_exp<<<enh_blocks, 256, 0, stream>>>(DST, ELOG, MAXB, SUMB, H);
        k_edge_alpha<<<enh_blocks, 256, 0, stream>>>(DST, ELOG, SUMB, H);
        // aggregate alpha * xl[src] into dst
        k_fill_u32<<<(unsigned)((MN + 255) / 256), 256, 0, stream>>>((unsigned*)AGG, 0u, MN);
        k_aggregate<<<ENED, (N >= 256 ? 256 : N), 0, stream>>>(SRC, DST, ELOG, XL, AGG, H);
        // bias + BN + ELU; f16 for next GEMM, f32 for layer-3 pooling input
        k_post<<<(unsigned)((MN + 255) / 256), 256, 0, stream>>>(
            AGG, bl[l], gl[l], bel[l], (l == 2) ? X3 : nullptr, XH, N, MN);
    }

    // pooling
    {
        size_t gc = (size_t)GG * CC;
        k_fill_u32<<<(unsigned)((gc + 255) / 256), 256, 0, stream>>>((unsigned*)PSUM, 0u, gc);
        k_fill_u32<<<(unsigned)((gc + 255) / 256), 256, 0, stream>>>(PMAX, FENC_NEG_INF, gc);
        k_fill_u32<<<1, 256, 0, stream>>>((unsigned*)PCNT, 0u, GG);
        k_pool_acc<<<NN, 128, 0, stream>>>(X3, batch, PSUM, PMAX, PCNT);
        k_pool_fin<<<(GG * CC + 255) / 256, 256, 0, stream>>>(PSUM, PMAX, PCNT, XG);
    }

    // heads
    k_head<<<GG, 160, 0, stream>>>(XG, pheno,
        (const float*)d_in[22], (const float*)d_in[23], (const float*)d_in[24], (const float*)d_in[25],
        (const float*)d_in[26], (const float*)d_in[27], (const float*)d_in[28], (const float*)d_in[29],
        (const float*)d_in[30], (const float*)d_in[31],
        (const float*)d_in[32], (const float*)d_in[33], (const float*)d_in[34], (const float*)d_in[35],
        (const float*)d_in[36], (const float*)d_in[37], (const float*)d_in[38], (const float*)d_in[39],
        (const float*)d_in[40], (const float*)d_in[41], (const float*)d_in[42], (const float*)d_in[43],
        out);
}
